// ScaledDotProductAttention_69750268887658
// MI455X (gfx1250) — matrix operand order repaired and checked by reading, not compiled, not run on hardware
//
#include <hip/hip_runtime.h>
#include <hip/hip_bf16.h>

typedef __attribute__((ext_vector_type(16))) _Float16 v16h;
typedef __attribute__((ext_vector_type(8)))  float    v8f;

// Problem constants from reference setup_inputs()
constexpr int B  = 16;
constexpr int L  = 2048;
constexpr int D  = 128;
constexpr int QT = 16;            // query rows per workgroup (one WMMA M tile)
constexpr int KT = 16;            // keys per WMMA N tile
constexpr int NWAVE = 8;          // waves per workgroup (256 threads, wave32)
constexpr int NTHREADS = NWAVE * 32;

// LDS: 16 x 2048 fp32 score/probability tile (128 KB) + padded reduce buffer.
// 320 KB per-WGP LDS on CDNA5 -> two workgroups per WGP still fit.
__global__ __launch_bounds__(NTHREADS)
void attn_fused_kernel(const float* __restrict__ Q,
                       const float* __restrict__ K,
                       const float* __restrict__ V,
                       const int*   __restrict__ M,
                       float* __restrict__ ctx,
                       float* __restrict__ pattn)
{
    __shared__ float S[QT * L];        // scores -> probabilities, row-major [QT][L]
    __shared__ float red[16 * 17];     // per-row partial max/sum, pad to 17

    const int tid  = threadIdx.x;
    const int wave = tid >> 5;
    const int lane = tid & 31;
    const int ln   = lane & 15;        // N/col index or M-row selector
    const int hi   = lane >> 4;        // half-wave selector per WMMA layouts

    const int qt = blockIdx.x;                 // global q-tile id
    const int b  = qt / (L / QT);
    const int q0 = (qt % (L / QT)) * QT;

    const float scale = rsqrtf((float)D);

    const float* Qb = Q + ((size_t)b * L + q0) * D;
    const float* Kb = K + (size_t)b * L * D;
    const float* Vb = V + (size_t)b * L * D;
    const int*   Mb = M + ((size_t)b * L + q0) * L;

    // ---- Q tile -> A fragments (f16 16x32, 4 chunks over D=128), scale folded in.
    // A order used here (contiguous, same as B): lane(ln) = row M; element e -> k = hi*16+e
    v16h afrag[4];
    {
        const float* qrow = Qb + (size_t)ln * D;
        #pragma unroll
        for (int c = 0; c < 4; ++c) {
            const int d0 = c * 32 + hi * 16;
            #pragma unroll
            for (int e = 0; e < 8; ++e) {
                afrag[c][e]     = (_Float16)(qrow[d0 + e]      * scale);
                afrag[c][e + 8] = (_Float16)(qrow[d0 + 8 + e] * scale);
            }
        }
    }

    // ---- Phase 1: scores = (Q*scale) @ K^T, masked, into LDS. Waves split key tiles.
    for (int j = wave; j < L / KT; j += NWAVE) {
        const int k0 = j * KT;

        // Hoist streaming (read-once) mask loads so they overlap the WMMA chain.
        int mk[8];
        #pragma unroll
        for (int r = 0; r < 8; ++r) {
            const int m = r + hi * 8;
            mk[r] = __builtin_nontemporal_load(&Mb[(size_t)m * L + (k0 + ln)]);
        }

        v8f acc = {};
        #pragma unroll
        for (int c = 0; c < 4; ++c) {
            // B (32x16) order used here (contiguous, same as A): lane(ln) = col N = key; element e -> row kk = hi*16+e
            // => contiguous 16 floats of K[key] starting at d = c*32 + hi*16
            v16h bfrag;
            const float* krow = Kb + (size_t)(k0 + ln) * D + c * 32 + hi * 16;
            #pragma unroll
            for (int e = 0; e < 16; ++e) bfrag[e] = (_Float16)krow[e];
            acc = __builtin_amdgcn_wmma_f32_16x16x32_f16(
                false, afrag[c], false, bfrag, (short)0, acc, false, false);
        }
        // C layout: VGPR r -> row m = r + hi*8, col n = ln. Apply mask, stash in LDS.
        #pragma unroll
        for (int r = 0; r < 8; ++r) {
            const int m = r + hi * 8;
            float s = acc[r];
            s = (mk[r] == 0) ? -1e9f : s;
            S[m * L + (k0 + ln)] = s;
        }
    }
    __syncthreads();

    // ---- Phase 2: row softmax in LDS. thread t: row = t&15, chunk = t>>4 (128 elems).
    {
        const int row = tid & 15;
        const int ch  = tid >> 4;
        float* srow = S + row * L + ch * (L / 16);

        float mx = -3.4e38f;
        #pragma unroll 8
        for (int i = 0; i < L / 16; ++i) mx = fmaxf(mx, srow[i]);
        red[row * 17 + ch] = mx;
        __syncthreads();
        float rmx = -3.4e38f;
        #pragma unroll
        for (int i = 0; i < 16; ++i) rmx = fmaxf(rmx, red[row * 17 + i]);
        __syncthreads();                 // all reads of maxes done before reuse

        float psum = 0.f;
        #pragma unroll 8
        for (int i = 0; i < L / 16; ++i) {
            const float e = __expf(srow[i] - rmx);
            srow[i] = e;
            psum += e;
        }
        red[row * 17 + ch] = psum;
        __syncthreads();
        float rsum = 0.f;
        #pragma unroll
        for (int i = 0; i < 16; ++i) rsum += red[row * 17 + i];
        const float inv = 1.f / rsum;
        #pragma unroll 8
        for (int i = 0; i < L / 16; ++i) srow[i] *= inv;
    }
    __syncthreads();

    // ---- Phase 3: write p_attn once, fully coalesced, non-temporal (write-once
    // 256 MB stream: keep it out of L2 so K/V stay resident there).
    {
        float* prow = pattn + ((size_t)b * L + q0) * L;
        #pragma unroll
        for (int r = 0; r < QT; ++r) {
            for (int c0 = tid; c0 < L; c0 += NTHREADS) {
                __builtin_nontemporal_store(S[r * L + c0], &prow[(size_t)r * L + c0]);
            }
        }
    }

    // ---- Phase 4: context = P @ V. Wave w owns d-slice [w*16, w*16+16).
    {
        const int n0 = wave * 16;
        v8f cacc = {};
        for (int kt = 0; kt < L / 32; ++kt) {
            const int k0 = kt * 32;
            // A fragment from LDS probabilities (row = ln), same 16-bit A layout.
            v16h pa;
            const float* prow_l = S + ln * L + k0 + hi * 16;
            #pragma unroll
            for (int e = 0; e < 8; ++e) {
                pa[e]     = (_Float16)prow_l[e];
                pa[e + 8] = (_Float16)prow_l[e + 8];
            }
            // B fragment: column d = n0+ln of V, rows k0+hi*16 .. +15.
            // Per-instruction this is coalesced: lanes 0-15 read 16 consecutive
            // floats of one V row (L2-resident).
            v16h vb;
            const float* vcol = Vb + (size_t)(k0 + hi * 16) * D + n0 + ln;
            #pragma unroll
            for (int e = 0; e < 16; ++e) vb[e] = (_Float16)vcol[(size_t)e * D];
            cacc = __builtin_amdgcn_wmma_f32_16x16x32_f16(
                false, pa, false, vb, (short)0, cacc, false, false);
        }
        // Store context: VGPR r -> row m = r + hi*8, col = n0 + ln. Write-once.
        float* crow = ctx + ((size_t)b * L + q0) * D;
        #pragma unroll
        for (int r = 0; r < 8; ++r) {
            __builtin_nontemporal_store(cacc[r],
                                        &crow[(size_t)(r + hi * 8) * D + n0 + ln]);
        }
    }
}

extern "C" void kernel_launch(void* const* d_in, const int* in_sizes, int n_in,
                              void* d_out, int out_size, void* d_ws, size_t ws_size,
                              hipStream_t stream) {
    (void)in_sizes; (void)n_in; (void)out_size; (void)d_ws; (void)ws_size;
    const float* Q = (const float*)d_in[0];
    const float* K = (const float*)d_in[1];
    const float* V = (const float*)d_in[2];
    const int*   M = (const int*)d_in[3];

    float* ctx   = (float*)d_out;                       // [B, L, D]
    float* pattn = ctx + (size_t)B * L * D;             // [B, L, L]

    const int nblocks = B * (L / QT);                   // 2048 q-tiles
    attn_fused_kernel<<<nblocks, NTHREADS, 0, stream>>>(Q, K, V, M, ctx, pattn);
}
